// TransformerBlock_48103633715846
// MI455X (gfx1250) — matrix operand order repaired and checked by reading, not compiled, not run on hardware
//
#include <hip/hip_runtime.h>

// ---------------------------------------------------------------------------
// GTrXL block for MI455X (gfx1250, wave32). All GEMMs via
// V_WMMA_F32_16X16X32_BF16 (bf16 operands, fp32 accumulate).
// LDS fragment loads vectorized to ds_load_b128; GEMM A-tile staged with
// CDNA5 async global->LDS loads (ASYNCcnt) + global_prefetch.
// ---------------------------------------------------------------------------

typedef __bf16 v16bf __attribute__((ext_vector_type(16)));
typedef float  v8f   __attribute__((ext_vector_type(8)));

#define DEV static __device__ __forceinline__

namespace {
constexpr int  kD  = 512;
constexpr int  kH  = 8;
constexpr int  kHD = 64;
constexpr int  kB  = 16;
constexpr int  kTQ = 512;
constexpr int  kTK = 1024;
constexpr long kMQ = (long)kB * kTQ;   // 8192 query rows
constexpr long kMK = (long)kB * kTK;   // 16384 key rows
}

DEV float gelu_exact(float x) {
  return 0.5f * x * (1.0f + erff(x * 0.70710678118654752440f));
}
DEV float sigmoid_f(float x) { return 1.0f / (1.0f + expf(-x)); }

DEV v8f wmma_bf16(v16bf a, v16bf b, v8f c) {
  return __builtin_amdgcn_wmma_f32_16x16x32_bf16(false, a, false, b,
                                                 (short)0, c, false, false);
}

union frag_u  { uint4 q[2]; v16bf v; __bf16 e[16]; };

// CDNA5 async copy: 16B global -> LDS, tracked by ASYNCcnt.
DEV void async_copy_b128(unsigned lds_off, const void* gaddr) {
  asm volatile("global_load_async_to_lds_b128 %0, %1, off"
               :: "v"(lds_off), "v"(gaddr) : "memory");
}
DEV void wait_asynccnt0() {
  asm volatile("s_wait_asynccnt 0x0" ::: "memory");
}
DEV unsigned lds_addr(const void* p) {   // LDS aperture: offset in low 32 bits
  return (unsigned)(size_t)p;
}

// A fragment: 16x32 bf16 from row-major [m][k] tile (row length `stride`).
// Per ISA layout each lane's 16 elements are two contiguous 8-elem runs:
//   K = ks + 8*hi + [0..7]  and  K = ks + 16 + 8*hi + [0..7]
DEV v16bf load_a_frag(const __bf16* base, int stride, int m0, int ks,
                      int lo, int hi) {
  frag_u u;
  const __bf16* p = base + (m0 + lo) * stride + ks + (hi << 3);
  u.q[0] = *(const uint4*)p;
  u.q[1] = *(const uint4*)(p + 16);
  return u.v;
}
// B fragment from [n][k] storage: lane lo -> column n, elements are the two
// 8-elem runs K = ks + 8*hi + [0..7] and K = ks + 16 + 8*hi + [0..7].
DEV v16bf load_b_frag_nk(const __bf16* base, int stride, int n0, int ks,
                         int lo, int hi) {
  frag_u u;
  const __bf16* p = base + (n0 + lo) * stride + ks + (hi << 3);
  u.q[0] = *(const uint4*)p;
  u.q[1] = *(const uint4*)(p + 16);
  return u.v;
}

// ---------------------------------------------------------------------------
// Generic batched GEMM: C[M,N] = act(optC + A[M,K] @ B[K,N] + bias)
// Batch z: b = z / Hdiv, h = z % Hdiv, offsets via (sXb, sXh) element strides.
// Block tile 64x64, K-step 64, 256 threads = 8 waves, 2 WMMA tiles per wave.
// A tile staged via async global->LDS; B tile staged transposed ([n][k]).
// ---------------------------------------------------------------------------
__global__ __launch_bounds__(256)
void gemm_bf16_kernel(const __bf16* __restrict__ A, long ldA, long sAb, long sAh,
                      const __bf16* __restrict__ B, long ldB, long sBb, long sBh,
                      float* __restrict__ Cf, __bf16* __restrict__ Cb,
                      long ldC, long sCb, long sCh,
                      const float* __restrict__ bias,
                      int M, int N, int K, int Hdiv, int act, int accum) {
  __shared__ __align__(16) __bf16 sA[64][64];    // [m][k], 8 KB
  __shared__ __align__(16) __bf16 sBt[64][64];   // [n][k], 8 KB

  const int t    = threadIdx.x;
  const int lane = t & 31;
  const int w    = t >> 5;
  const int lo   = lane & 15;
  const int hi   = lane >> 4;

  const int n0 = blockIdx.x * 64;
  const int m0 = blockIdx.y * 64;
  const int zb = (int)blockIdx.z / Hdiv;
  const int zh = (int)blockIdx.z % Hdiv;

  const __bf16* Ab = A + (long)zb * sAb + (long)zh * sAh;
  const __bf16* Bb = B + (long)zb * sBb + (long)zh * sBh;

  const int tm = w & 3;            // wave's M tile (0..3)
  const int tn = (w >> 2) << 1;    // wave's first N tile (0 or 2)

  v8f acc0 = {0.f, 0.f, 0.f, 0.f, 0.f, 0.f, 0.f, 0.f};
  v8f acc1 = acc0;

  const int arow = t >> 2, acol = (t & 3) * 16;  // 64x64 A tile, 16 bf16/thread
  const int brow = t >> 2, bcol = (t & 3) * 16;  // 64x64 B tile

  const unsigned ldsA0 = lds_addr(&sA[arow][acol]);
  const unsigned ldsA1 = lds_addr(&sA[arow][acol + 8]);

  for (int k0 = 0; k0 < K; k0 += 64) {
    // A tile: direct async copy (same layout in LDS as in memory).
    const __bf16* ag = Ab + (long)(m0 + arow) * ldA + k0 + acol;
    async_copy_b128(ldsA0, ag);
    async_copy_b128(ldsA1, ag + 8);

    // B tile: load 16 bf16 through VGPRs, store transposed into sBt[n][k].
    const __bf16* bg = Bb + (long)(k0 + brow) * ldB + n0 + bcol;
    frag_u pb;
    pb.q[0] = *(const uint4*)bg;
    pb.q[1] = *(const uint4*)(bg + 8);

    if (k0 + 64 < K) {  // prefetch next K tile (global_prefetch_b8)
      __builtin_prefetch(ag + 64, 0, 1);
      __builtin_prefetch(bg + 64 * ldB, 0, 1);
    }

#pragma unroll
    for (int j = 0; j < 16; ++j) {
      int jj = (j + t) & 15;                    // stagger LDS banks
      sBt[bcol + jj][brow] = pb.e[jj];
    }

    wait_asynccnt0();
    __syncthreads();

    v16bf a0  = load_a_frag(&sA[0][0], 64, tm << 4, 0,  lo, hi);
    v16bf a1  = load_a_frag(&sA[0][0], 64, tm << 4, 32, lo, hi);
    v16bf b00 = load_b_frag_nk(&sBt[0][0], 64, tn << 4, 0,  lo, hi);
    v16bf b01 = load_b_frag_nk(&sBt[0][0], 64, tn << 4, 32, lo, hi);
    v16bf b10 = load_b_frag_nk(&sBt[0][0], 64, (tn + 1) << 4, 0,  lo, hi);
    v16bf b11 = load_b_frag_nk(&sBt[0][0], 64, (tn + 1) << 4, 32, lo, hi);
    acc0 = wmma_bf16(a0, b00, acc0);
    acc0 = wmma_bf16(a1, b01, acc0);
    acc1 = wmma_bf16(a0, b10, acc1);
    acc1 = wmma_bf16(a1, b11, acc1);
    __syncthreads();
  }

  const long cbase = (long)zb * sCb + (long)zh * sCh;
#pragma unroll
  for (int i = 0; i < 2; ++i) {
    v8f acc  = i ? acc1 : acc0;
    int col  = n0 + ((tn + i) << 4) + lo;
    float bv = bias ? bias[col] : 0.0f;
#pragma unroll
    for (int r = 0; r < 8; ++r) {
      int  row = m0 + (tm << 4) + (hi << 3) + r;
      long idx = cbase + (long)row * ldC + col;
      float v  = acc[r] + bv;
      if (accum) v += Cf[idx];
      if (act == 1) v = gelu_exact(v);
      if (Cf) Cf[idx] = v;
      if (Cb) Cb[idx] = (__bf16)v;
    }
  }
}

// ---------------------------------------------------------------------------
// Fused Transformer-XL score tile:
//   scores[b,h,i,j] = (AC[i,j] + BDraw[i, j+TQ-1-i]) / sqrt(HD)
// Per block: one (b,h, 64q x 64k) tile. AC via WMMA against k^T; the 128-wide
// position band via WMMA against p^T; rel-shift gather in the epilogue.
// Zero band rows beyond TK realize the rel-shift zero padding.
// ---------------------------------------------------------------------------
__global__ __launch_bounds__(256)
void scores_kernel(const __bf16* __restrict__ qU, const __bf16* __restrict__ qV,
                   const __bf16* __restrict__ Kb, const __bf16* __restrict__ P,
                   float* __restrict__ S) {
  __shared__ __align__(16) char smem[64 * 64 * 4 + 64 * 128 * 4];  // 48 KB
  __bf16 (*sQU)[64] = (__bf16(*)[64])(smem);                 //  8 KB [m][d]
  __bf16 (*sQV)[64] = (__bf16(*)[64])(smem + 8192);          //  8 KB [m][d]
  __bf16 (*sK)[64]  = (__bf16(*)[64])(smem + 16384);         //  8 KB [j][d]
  __bf16 (*sP)[64]  = (__bf16(*)[64])(smem + 24576);         // 16 KB [c][d]
  float (*aAC)[64]  = (float(*)[64])(smem);                  // overlay, 16 KB
  float (*aBD)[128] = (float(*)[128])(smem + 16384);         // overlay, 32 KB

  const int t    = threadIdx.x;
  const int lane = t & 31;
  const int w    = t >> 5;
  const int lo   = lane & 15;
  const int hi   = lane >> 4;

  const int j0 = blockIdx.x * 64;
  const int i0 = blockIdx.y * 64;
  const int bz = blockIdx.z;           // b*H + h
  const int b  = bz >> 3;
  const int h  = bz & 7;

  const long qoff = ((long)b * kTQ + i0) * kD + (long)h * kHD;
  const long koff = ((long)b * kTK + j0) * kD + (long)h * kHD;

  {  // stage qU, qV, K tiles: 64 rows x 64 d
    int row = t >> 2, c0 = (t & 3) * 16;
    *(uint4*)&sQU[row][c0]     = *(const uint4*)(qU + qoff + (long)row * kD + c0);
    *(uint4*)&sQU[row][c0 + 8] = *(const uint4*)(qU + qoff + (long)row * kD + c0 + 8);
    *(uint4*)&sQV[row][c0]     = *(const uint4*)(qV + qoff + (long)row * kD + c0);
    *(uint4*)&sQV[row][c0 + 8] = *(const uint4*)(qV + qoff + (long)row * kD + c0 + 8);
    *(uint4*)&sK[row][c0]      = *(const uint4*)(Kb + koff + (long)row * kD + c0);
    *(uint4*)&sK[row][c0 + 8]  = *(const uint4*)(Kb + koff + (long)row * kD + c0 + 8);
  }
  {  // stage position band: rows c in [0,128), p row = j0 - i0 + (TQ-64) + c
    int c  = t >> 1;
    int c0 = (t & 1) * 32;
    int pr = j0 - i0 + (kTQ - 64) + c;
    if (pr < kTK) {
      const __bf16* src = P + (long)pr * kD + (long)h * kHD + c0;
#pragma unroll
      for (int j = 0; j < 4; ++j)
        *(uint4*)&sP[c][c0 + 8 * j] = *(const uint4*)(src + 8 * j);
    } else {
      uint4 z4 = make_uint4(0u, 0u, 0u, 0u);
#pragma unroll
      for (int j = 0; j < 4; ++j) *(uint4*)&sP[c][c0 + 8 * j] = z4;
    }
  }
  __syncthreads();

  // AC (4x4 tiles): wave w owns (tmA, tnA), (tmA+1, tnA).
  // BD band (4x8 tiles): wave w owns column tile w, rows 0..3.
  const int tmA = (2 * w) & 3;
  const int tnA = w >> 1;

  v8f z8 = {0.f, 0.f, 0.f, 0.f, 0.f, 0.f, 0.f, 0.f};
  v8f ac0 = z8, ac1 = z8;
  v8f bd[4] = {z8, z8, z8, z8};

#pragma unroll
  for (int ks = 0; ks < 64; ks += 32) {
    v16bf bk = load_b_frag_nk(&sK[0][0], 64, tnA << 4, ks, lo, hi);
    ac0 = wmma_bf16(load_a_frag(&sQU[0][0], 64, tmA << 4, ks, lo, hi), bk, ac0);
    ac1 = wmma_bf16(load_a_frag(&sQU[0][0], 64, (tmA + 1) << 4, ks, lo, hi), bk, ac1);
    v16bf bp = load_b_frag_nk(&sP[0][0], 64, w << 4, ks, lo, hi);
#pragma unroll
    for (int r4 = 0; r4 < 4; ++r4)
      bd[r4] = wmma_bf16(load_a_frag(&sQV[0][0], 64, r4 << 4, ks, lo, hi), bp, bd[r4]);
  }
  __syncthreads();  // operand LDS dead; overlay becomes accumulators

#pragma unroll
  for (int r = 0; r < 8; ++r) {
    aAC[(tmA << 4) + (hi << 3) + r][(tnA << 4) + lo]       = ac0[r];
    aAC[((tmA + 1) << 4) + (hi << 3) + r][(tnA << 4) + lo] = ac1[r];
  }
#pragma unroll
  for (int r4 = 0; r4 < 4; ++r4)
#pragma unroll
    for (int r = 0; r < 8; ++r)
      aBD[(r4 << 4) + (hi << 3) + r][(w << 4) + lo] = bd[r4][r];
  __syncthreads();

  const int di  = t >> 2;
  const int dj0 = (t & 3) * 16;
  const long srow = ((long)bz * kTQ + i0 + di) * kTK + j0;
#pragma unroll
  for (int v4 = 0; v4 < 4; ++v4) {
    float4 o;
    int dj = dj0 + v4 * 4;
    o.x = (aAC[di][dj + 0] + aBD[di][dj + 63 - di + 0]) * 0.125f;
    o.y = (aAC[di][dj + 1] + aBD[di][dj + 63 - di + 1]) * 0.125f;
    o.z = (aAC[di][dj + 2] + aBD[di][dj + 63 - di + 2]) * 0.125f;
    o.w = (aAC[di][dj + 3] + aBD[di][dj + 63 - di + 3]) * 0.125f;
    *(float4*)&S[srow + dj] = o;
  }
}

// ---------------------------------------------------------------------------
// Row softmax over TK=1024, fp32 in -> bf16 probabilities. Wave per row.
// ---------------------------------------------------------------------------
__global__ __launch_bounds__(256)
void softmax_kernel(const float* __restrict__ S, __bf16* __restrict__ Pr) {
  const int lane = threadIdx.x & 31;
  const int w    = threadIdx.x >> 5;
  const long row = (long)blockIdx.x * 8 + w;
  const float* sr = S + row * kTK;
  float v[32];
  float mx = -3.4e38f;
#pragma unroll
  for (int i = 0; i < 32; ++i) {
    v[i] = sr[i * 32 + lane];
    mx   = fmaxf(mx, v[i]);
  }
#pragma unroll
  for (int m = 16; m; m >>= 1) mx = fmaxf(mx, __shfl_xor(mx, m, 32));
  float sum = 0.f;
#pragma unroll
  for (int i = 0; i < 32; ++i) {
    v[i] = expf(v[i] - mx);
    sum += v[i];
  }
#pragma unroll
  for (int m = 16; m; m >>= 1) sum += __shfl_xor(sum, m, 32);
  const float inv = 1.0f / sum;
  __bf16* pr = Pr + row * kTK;
#pragma unroll
  for (int i = 0; i < 32; ++i) pr[i * 32 + lane] = (__bf16)(v[i] * inv);
}

// ---------------------------------------------------------------------------
// LayerNorm over 512 columns, fp32 in -> bf16 out. Wave per row.
// ---------------------------------------------------------------------------
__global__ __launch_bounds__(256)
void ln_kernel(const float* __restrict__ X, const float* __restrict__ g,
               const float* __restrict__ be, __bf16* __restrict__ O) {
  const int lane = threadIdx.x & 31;
  const int w    = threadIdx.x >> 5;
  const long row = (long)blockIdx.x * 8 + w;
  const float* xr = X + row * kD;
  float v[16], s = 0.f, s2 = 0.f;
#pragma unroll
  for (int i = 0; i < 16; ++i) {
    float x = xr[i * 32 + lane];
    v[i] = x; s += x; s2 += x * x;
  }
#pragma unroll
  for (int m = 16; m; m >>= 1) {
    s  += __shfl_xor(s, m, 32);
    s2 += __shfl_xor(s2, m, 32);
  }
  const float mu   = s * (1.0f / kD);
  const float var  = s2 * (1.0f / kD) - mu * mu;
  const float rinv = rsqrtf(var + 1e-5f);
#pragma unroll
  for (int i = 0; i < 16; ++i) {
    int c = i * 32 + lane;
    O[row * kD + c] = (__bf16)((v[i] - mu) * rinv * g[c] + be[c]);
  }
}

// ---------------------------------------------------------------------------
// Elementwise helpers
// ---------------------------------------------------------------------------
__global__ __launch_bounds__(256)
void cast_bf16_kernel(const float* __restrict__ in, __bf16* __restrict__ out, long n) {
  long i = (long)blockIdx.x * blockDim.x + threadIdx.x;
  if (i < n) out[i] = (__bf16)in[i];
}

__global__ __launch_bounds__(256)
void qbias_kernel(const float* __restrict__ q, const float* __restrict__ U,
                  const float* __restrict__ V, __bf16* __restrict__ qU,
                  __bf16* __restrict__ qV, long n) {
  long i = (long)blockIdx.x * blockDim.x + threadIdx.x;
  if (i < n) {
    int c   = (int)(i & (kD - 1));  // column == flat (h,hd) index
    float x = q[i];
    qU[i] = (__bf16)(x + U[c]);
    qV[i] = (__bf16)(x + V[c]);
  }
}

__global__ __launch_bounds__(256)
void gate_rx_kernel(const float* __restrict__ Gr, const float* __restrict__ x,
                    __bf16* __restrict__ rx, long n) {
  long i = (long)blockIdx.x * blockDim.x + threadIdx.x;
  if (i < n) rx[i] = (__bf16)(sigmoid_f(Gr[i]) * x[i]);
}

__global__ __launch_bounds__(256)
void gate_out_kernel(const float* __restrict__ Gz, const float* __restrict__ Gg,
                     const float* __restrict__ x, const float* __restrict__ bg,
                     float* __restrict__ outf, __bf16* __restrict__ outb, long n) {
  long i = (long)blockIdx.x * blockDim.x + threadIdx.x;
  if (i < n) {
    int c    = (int)(i & (kD - 1));
    float z  = sigmoid_f(Gz[i] - bg[c]);
    float hh = tanhf(Gg[i]);
    float o  = (1.0f - z) * x[i] + z * hh;
    outf[i] = o;
    if (outb) outb[i] = (__bf16)o;
  }
}

// ---------------------------------------------------------------------------
// Host orchestration
// ---------------------------------------------------------------------------
extern "C" void kernel_launch(void* const* d_in, const int* in_sizes, int n_in,
                              void* d_out, int out_size, void* d_ws, size_t ws_size,
                              hipStream_t stream) {
  (void)in_sizes; (void)n_in; (void)out_size; (void)ws_size;

  const float* query = (const float*)d_in[0];
  const float* keyi  = (const float*)d_in[1];
  const float* pos   = (const float*)d_in[2];
  const float* U     = (const float*)d_in[3];
  const float* V     = (const float*)d_in[4];
  const float* attnW = (const float*)d_in[5];
  const float* ln1g  = (const float*)d_in[6];
  const float* ln1b  = (const float*)d_in[7];
  const float* ln2g  = (const float*)d_in[8];
  const float* ln2b  = (const float*)d_in[9];
  const float* g1W   = (const float*)d_in[10];
  const float* g1b   = (const float*)d_in[11];
  const float* g2W   = (const float*)d_in[12];
  const float* g2b   = (const float*)d_in[13];
  const float* fcW1  = (const float*)d_in[14];
  const float* fcb1  = (const float*)d_in[15];
  const float* fcW2  = (const float*)d_in[16];
  const float* fcb2  = (const float*)d_in[17];

  const long DD = (long)kD * kD;
  const long nQ = kMQ * kD;          // 4M elems
  const long nK = kMK * kD;          // 8M elems

  char*  ws  = (char*)d_ws;
  size_t off = 0;
  auto alloc = [&](size_t bytes) -> void* {
    void* p = ws + off;
    off += (bytes + 255) & ~(size_t)255;
    return p;
  };

  __bf16* nq_bf   = (__bf16*)alloc(nQ * 2);
  __bf16* nk_bf   = (__bf16*)alloc(nK * 2);
  __bf16* pos_bf  = (__bf16*)alloc((long)kTK * kD * 2);
  __bf16* aW_bf   = (__bf16*)alloc(5 * DD * 2);
  __bf16* g1W_bf  = (__bf16*)alloc(6 * DD * 2);
  __bf16* g2W_bf  = (__bf16*)alloc(6 * DD * 2);
  __bf16* fc1_bf  = (__bf16*)alloc((long)kD * 4 * kD * 2);
  __bf16* fc2_bf  = (__bf16*)alloc((long)kD * 4 * kD * 2);
  __bf16* xq_bf   = (__bf16*)alloc(nQ * 2);
  float*  qbuf    = (float*)alloc(nQ * 4);
  __bf16* qU_bf   = (__bf16*)alloc(nQ * 2);
  __bf16* qV_bf   = (__bf16*)alloc(nQ * 2);
  __bf16* k_bf    = (__bf16*)alloc(nK * 2);
  __bf16* v_bf    = (__bf16*)alloc(nK * 2);
  __bf16* pp_bf   = (__bf16*)alloc((long)kTK * kD * 2);
  float*  scores  = (float*)alloc((long)kB * kH * kTQ * kTK * 4);   // 256 MB
  __bf16* attn_bf = (__bf16*)alloc((long)kB * kH * kTQ * kTK * 2);  // 128 MB
  __bf16* yh_bf   = (__bf16*)alloc(nQ * 2);
  __bf16* y1_bf   = (__bf16*)alloc(nQ * 2);
  float*  Gr      = (float*)alloc(nQ * 4);
  float*  Gz      = (float*)alloc(nQ * 4);
  float*  Gg      = (float*)alloc(nQ * 4);
  __bf16* rx_bf   = (__bf16*)alloc(nQ * 2);
  float*  out1    = (float*)alloc(nQ * 4);
  __bf16* out1_bf = (__bf16*)alloc(nQ * 2);
  __bf16* h2_bf   = (__bf16*)alloc(nQ * 2);
  __bf16* T1_bf   = (__bf16*)alloc(kMQ * 4 * kD * 2);
  __bf16* E_bf    = (__bf16*)alloc(nQ * 2);

  const __bf16 *Wq = aW_bf, *Wk = aW_bf + DD, *Wv = aW_bf + 2 * DD,
               *Wp = aW_bf + 3 * DD, *Wo = aW_bf + 4 * DD;

  auto cast = [&](const float* in, __bf16* o, long n) {
    cast_bf16_kernel<<<dim3((unsigned)((n + 255) / 256)), dim3(256), 0, stream>>>(in, o, n);
  };
  auto gemm = [&](const __bf16* A, long ldA, long sAb, long sAh,
                  const __bf16* Bm, long ldB, long sBb, long sBh,
                  float* Cf, __bf16* Cb, long ldC, long sCb, long sCh,
                  const float* bias, long M, long N, long K,
                  int Z, int Hdiv, int act, int accum) {
    dim3 g((unsigned)(N / 64), (unsigned)(M / 64), (unsigned)Z);
    gemm_bf16_kernel<<<g, dim3(256), 0, stream>>>(A, ldA, sAb, sAh, Bm, ldB, sBb, sBh,
                                                  Cf, Cb, ldC, sCb, sCh, bias,
                                                  (int)M, (int)N, (int)K, Hdiv, act, accum);
  };

  // --- precision conversion of weights / raw activations ---
  cast(pos, pos_bf, (long)kTK * kD);
  cast(attnW, aW_bf, 5 * DD);
  cast(g1W, g1W_bf, 6 * DD);
  cast(g2W, g2W_bf, 6 * DD);
  cast(fcW1, fc1_bf, (long)kD * 4 * kD);
  cast(fcW2, fc2_bf, (long)kD * 4 * kD);
  cast(query, xq_bf, nQ);

  // --- LayerNorm 1 on query and key ---
  ln_kernel<<<dim3((unsigned)(kMQ / 8)), dim3(256), 0, stream>>>(query, ln1g, ln1b, nq_bf);
  ln_kernel<<<dim3((unsigned)(kMK / 8)), dim3(256), 0, stream>>>(keyi, ln1g, ln1b, nk_bf);

  // --- projections ---
  gemm(nq_bf, kD, 0, 0, Wq, kD, 0, 0, qbuf, nullptr, kD, 0, 0, nullptr, kMQ, kD, kD, 1, 1, 0, 0);
  qbias_kernel<<<dim3((unsigned)(nQ / 256)), dim3(256), 0, stream>>>(qbuf, U, V, qU_bf, qV_bf, nQ);
  gemm(nk_bf, kD, 0, 0, Wk, kD, 0, 0, nullptr, k_bf, kD, 0, 0, nullptr, kMK, kD, kD, 1, 1, 0, 0);
  gemm(nk_bf, kD, 0, 0, Wv, kD, 0, 0, nullptr, v_bf, kD, 0, 0, nullptr, kMK, kD, kD, 1, 1, 0, 0);
  gemm(pos_bf, kD, 0, 0, Wp, kD, 0, 0, nullptr, pp_bf, kD, 0, 0, nullptr, kTK, kD, kD, 1, 1, 0, 0);

  // --- attention scores (AC + rel-shifted BD), softmax, attn @ V ---
  scores_kernel<<<dim3(kTK / 64, kTQ / 64, kB * kH), dim3(256), 0, stream>>>(
      qU_bf, qV_bf, k_bf, pp_bf, scores);
  softmax_kernel<<<dim3((unsigned)((long)kB * kH * kTQ / 8)), dim3(256), 0, stream>>>(
      scores, attn_bf);
  gemm(attn_bf, kTK, (long)kH * kTQ * kTK, (long)kTQ * kTK,
       v_bf, kD, (long)kTK * kD, kHD,
       nullptr, yh_bf, kD, (long)kTQ * kD, kHD,
       nullptr, kTQ, kHD, kTK, kB * kH, kH, 0, 0);

  // --- output projection + exact GELU ---
  gemm(yh_bf, kD, 0, 0, Wo, kD, 0, 0, nullptr, y1_bf, kD, 0, 0, nullptr, kMQ, kD, kD, 1, 1, 1, 0);

  // --- GRU gate 1: x = query, y = y1 ---
  const __bf16 *g1Wr = g1W_bf, *g1Ur = g1W_bf + DD, *g1Wz = g1W_bf + 2 * DD,
               *g1Uz = g1W_bf + 3 * DD, *g1Wg = g1W_bf + 4 * DD, *g1Ug = g1W_bf + 5 * DD;
  gemm(y1_bf, kD, 0, 0, g1Wr, kD, 0, 0, Gr, nullptr, kD, 0, 0, nullptr, kMQ, kD, kD, 1, 1, 0, 0);
  gemm(xq_bf, kD, 0, 0, g1Ur, kD, 0, 0, Gr, nullptr, kD, 0, 0, nullptr, kMQ, kD, kD, 1, 1, 0, 1);
  gemm(y1_bf, kD, 0, 0, g1Wz, kD, 0, 0, Gz, nullptr, kD, 0, 0, nullptr, kMQ, kD, kD, 1, 1, 0, 0);
  gemm(xq_bf, kD, 0, 0, g1Uz, kD, 0, 0, Gz, nullptr, kD, 0, 0, nullptr, kMQ, kD, kD, 1, 1, 0, 1);
  gemm(y1_bf, kD, 0, 0, g1Wg, kD, 0, 0, Gg, nullptr, kD, 0, 0, nullptr, kMQ, kD, kD, 1, 1, 0, 0);
  gate_rx_kernel<<<dim3((unsigned)(nQ / 256)), dim3(256), 0, stream>>>(Gr, query, rx_bf, nQ);
  gemm(rx_bf, kD, 0, 0, g1Ug, kD, 0, 0, Gg, nullptr, kD, 0, 0, nullptr, kMQ, kD, kD, 1, 1, 0, 1);
  gate_out_kernel<<<dim3((unsigned)(nQ / 256)), dim3(256), 0, stream>>>(
      Gz, Gg, query, g1b, out1, out1_bf, nQ);

  // --- LN2 + MLP (GELU(GELU(h2 W1 + b1) W2 + b2)) ---
  ln_kernel<<<dim3((unsigned)(kMQ / 8)), dim3(256), 0, stream>>>(out1, ln2g, ln2b, h2_bf);
  gemm(h2_bf, kD, 0, 0, fc1_bf, 4 * kD, 0, 0, nullptr, T1_bf, 4 * kD, 0, 0, fcb1,
       kMQ, 4 * kD, kD, 1, 1, 1, 0);
  gemm(T1_bf, 4 * kD, 0, 0, fc2_bf, kD, 0, 0, nullptr, E_bf, kD, 0, 0, fcb2,
       kMQ, kD, 4 * kD, 1, 1, 1, 0);

  // --- GRU gate 2: x = out1, y = E; final output fp32 to d_out ---
  const __bf16 *g2Wr = g2W_bf, *g2Ur = g2W_bf + DD, *g2Wz = g2W_bf + 2 * DD,
               *g2Uz = g2W_bf + 3 * DD, *g2Wg = g2W_bf + 4 * DD, *g2Ug = g2W_bf + 5 * DD;
  gemm(E_bf, kD, 0, 0, g2Wr, kD, 0, 0, Gr, nullptr, kD, 0, 0, nullptr, kMQ, kD, kD, 1, 1, 0, 0);
  gemm(out1_bf, kD, 0, 0, g2Ur, kD, 0, 0, Gr, nullptr, kD, 0, 0, nullptr, kMQ, kD, kD, 1, 1, 0, 1);
  gemm(E_bf, kD, 0, 0, g2Wz, kD, 0, 0, Gz, nullptr, kD, 0, 0, nullptr, kMQ, kD, kD, 1, 1, 0, 0);
  gemm(out1_bf, kD, 0, 0, g2Uz, kD, 0, 0, Gz, nullptr, kD, 0, 0, nullptr, kMQ, kD, kD, 1, 1, 0, 1);
  gemm(E_bf, kD, 0, 0, g2Wg, kD, 0, 0, Gg, nullptr, kD, 0, 0, nullptr, kMQ, kD, kD, 1, 1, 0, 0);
  gate_rx_kernel<<<dim3((unsigned)(nQ / 256)), dim3(256), 0, stream>>>(Gr, out1, rx_bf, nQ);
  gemm(rx_bf, kD, 0, 0, g2Ug, kD, 0, 0, Gg, nullptr, kD, 0, 0, nullptr, kMQ, kD, kD, 1, 1, 0, 1);
  gate_out_kernel<<<dim3((unsigned)(nQ / 256)), dim3(256), 0, stream>>>(
      Gz, Gg, out1, g2b, (float*)d_out, nullptr, nQ);
}
